// TypeAreaMultiHeadAttention_36507222016298
// MI455X (gfx1250) — compile-verified, matrix operand order checked by reading, not run on hardware
//
#include <hip/hip_runtime.h>
#include <hip/hip_bf16.h>

typedef __attribute__((ext_vector_type(16))) _Float16 v16h;
typedef __attribute__((ext_vector_type(8)))  _Float16 v8h;
typedef __attribute__((ext_vector_type(4)))  _Float16 v4h;
typedef __attribute__((ext_vector_type(8)))  float    v8f;

#define BB 4
#define LL 2048
#define DD 1024
#define HH 16
#define DKV 64
#define M_TOT (BB * LL)           // 8192
#define SZA ((size_t)M_TOT * DD)  // one activation tensor, elements
#define SZW ((size_t)DD * DD)     // one weight, elements
#define SZQ ((size_t)BB * HH * LL * DKV)
#define INV_SCALE 0.0625f         // 1/sqrt(4*64)

#if defined(__gfx1250__) && __has_builtin(__builtin_amdgcn_tensor_load_to_lds) && \
    __has_builtin(__builtin_amdgcn_s_wait_tensorcnt)
#define HAVE_TDM 1
#else
#define HAVE_TDM 0
#endif

// ---------------------------------------------------------------------------
// WMMA helpers
// ---------------------------------------------------------------------------
__device__ __forceinline__ v8f wmma16(v16h a, v16h b, v8f c) {
  return __builtin_amdgcn_wmma_f32_16x16x32_f16(
      /*neg_a=*/false, a, /*neg_b=*/false, b,
      /*c_mod=*/(short)0, c, /*reuse_a=*/false, /*reuse_b=*/false);
}

// Load a 16x32 f16 fragment whose "rows" (lane&15) are contiguous in k.
// Per ISA 16-bit A-matrix table: lane<16 holds k=[0..7],[16..23],
// lane>=16 holds k=[8..15],[24..31]; two 16B loads per lane.
__device__ __forceinline__ v16h load_frag16(const _Float16* __restrict__ p0,
                                            int stride, int lane) {
  const int r  = lane & 15;
  const int kh = (lane >> 4) << 3;  // 0 or 8
  const _Float16* p = p0 + (size_t)r * stride + kh;
  v8h lo = *(const v8h*)(p);
  v8h hi = *(const v8h*)(p + 16);
  v16h o;
#pragma unroll
  for (int i = 0; i < 8; ++i) { o[i] = lo[i]; o[i + 8] = hi[i]; }
  return o;
}

#if HAVE_TDM
typedef __attribute__((ext_vector_type(4))) unsigned int u32x4;
typedef __attribute__((ext_vector_type(8))) int          i32x8;
typedef __attribute__((ext_vector_type(4))) int          i32x4;

// Issue one TDM 2-D tile load: rows x cols f16 elements, global row stride
// `stride_e` elements, into LDS at byte offset lds_off. D# packing follows
// CDNA5 ISA (group0: count/lds/global/type,
// group1: data_size, tensor dims, tile dims, dim0 stride). This toolchain's
// builtin takes 6 args: (g0, g1, g2, g3, extra, cpol).
__device__ __forceinline__ void tdm_load_2d(unsigned long long gaddr,
                                            unsigned int lds_off,
                                            unsigned int rows, unsigned int cols,
                                            unsigned long long stride_e) {
  u32x4 g0;
  g0[0] = 1u;                                            // count=1 (valid), user mode
  g0[1] = lds_off;                                       // lds_addr (bytes)
  g0[2] = (unsigned int)(gaddr & 0xffffffffu);           // global_addr[31:0]
  g0[3] = (unsigned int)((gaddr >> 32) & 0x01ffffffu)    // global_addr[56:32]
          | (2u << 30);                                  // type=2 ("image")
  i32x8 g1;
  g1[0] = (int)(1u << 16);                               // data_size=1 (2 bytes)
  g1[1] = (int)((cols & 0xffffu) << 16);                 // tensor_dim0[15:0]
  g1[2] = (int)(((rows & 0xffffu) << 16) | (cols >> 16));// tensor_dim1 lo | dim0 hi
  g1[3] = (int)(((cols & 0xffffu) << 16) | (rows >> 16));// tile_dim0 | tensor_dim1 hi
  g1[4] = (int)(rows & 0xffffu);                         // tile_dim1 (tile_dim2=0)
  g1[5] = (int)(stride_e & 0xffffffffu);                 // tensor_dim0_stride lo
  g1[6] = (int)((stride_e >> 32) & 0xffffu);             // stride hi (dim1_stride=0)
  g1[7] = 0;
  const i32x4 z4 = {0, 0, 0, 0};
  const i32x8 z8 = {0, 0, 0, 0, 0, 0, 0, 0};
  __builtin_amdgcn_tensor_load_to_lds(g0, g1, z4, z4, z8, 0);
}
#endif

// ---------------------------------------------------------------------------
// 1) fp32 -> f16 activation conversion (vector x4)
// ---------------------------------------------------------------------------
__global__ __launch_bounds__(256) void cvt_act_kernel(const float* __restrict__ in,
                                                      _Float16* __restrict__ out) {
  size_t i = ((size_t)blockIdx.x * blockDim.x + threadIdx.x) * 4;
  float4 v = *(const float4*)(in + i);
  v4h o = {(_Float16)v.x, (_Float16)v.y, (_Float16)v.z, (_Float16)v.w};
  *(v4h*)(out + i) = o;
}

// ---------------------------------------------------------------------------
// 2) fp32 weight -> f16 transposed ([in,out] -> [out,in]) via LDS tile
// ---------------------------------------------------------------------------
__global__ __launch_bounds__(256) void cvt_w_t_kernel(const float* __restrict__ W,
                                                      _Float16* __restrict__ Wt) {
  __shared__ float tile[32][33];
  const int i0 = blockIdx.x * 32;  // input-row base
  const int o0 = blockIdx.y * 32;  // input-col base (= output-row base)
#pragma unroll
  for (int rr = threadIdx.y; rr < 32; rr += 8)
    tile[rr][threadIdx.x] = W[(size_t)(i0 + rr) * DD + o0 + threadIdx.x];
  __syncthreads();
#pragma unroll
  for (int rr = threadIdx.y; rr < 32; rr += 8)
    Wt[(size_t)(o0 + rr) * DD + i0 + threadIdx.x] = (_Float16)tile[threadIdx.x][rr];
}

// ---------------------------------------------------------------------------
// 3) QKV projection GEMM:  out = act_a @ W_a + act_s @ W_s  (K = 2x1024)
//    grid (M/128, N/64, 3); block 128 (4 waves); wave tile = 32M x 64N
//    Weight (B) tiles staged in LDS via TDM, double buffered, TENSORcnt-
//    pipelined so the DMA of tile k+1 overlaps WMMA on tile k.
//    A/W selected by integer offset (not pointer select) so addrspace
//    inference keeps A loads as global_load_b128, not flat.
//    z=0 -> Q [B,H,L,64]; z=1 -> K [B,H,L,64]; z=2 -> V^T [B,H,64,L]
// ---------------------------------------------------------------------------
__global__ __launch_bounds__(128) void qkv_gemm_kernel(
    const _Float16* __restrict__ act, const _Float16* __restrict__ wt,
    _Float16* __restrict__ Qo, _Float16* __restrict__ Ko,
    _Float16* __restrict__ Vt) {
  __shared__ _Float16 bbuf[2][64 * 32];  // double-buffered 64n x 32k tile (4KB ea)

  const int z    = blockIdx.z;
  const int bm   = blockIdx.x * 128;
  const int bn   = blockIdx.y * 64;
  const int wave = threadIdx.x >> 5;
  const int lane = threadIdx.x & 31;
  const int m0   = bm + wave * 32;

  const _Float16* Abase = act + (size_t)z * SZA;  // second tensor at +3*SZA
  const _Float16* Wbase = wt + (size_t)z * SZW;   // second weight at +3*SZW

  v8f acc[2][4] = {};

  auto stage = [&](int it) {  // it in [0,64): p = it>>5, k0 = (it&31)*32
    const size_t poff = (it >> 5) ? (size_t)3 * SZW : (size_t)0;  // int select
    const int k0 = (it & 31) << 5;
    const _Float16* src = Wbase + poff + (size_t)bn * DD + k0;
    _Float16* dst = &bbuf[it & 1][0];
#if HAVE_TDM
    if (wave == 0)
      tdm_load_2d((unsigned long long)(size_t)src,
                  (unsigned int)(size_t)dst, /*rows=*/64, /*cols=*/32,
                  /*stride=*/DD);
#else
    const int c0 = threadIdx.x * 2;  // 16 halves per thread, 2 x 16B chunks
#pragma unroll
    for (int c = c0; c < c0 + 2; ++c) {
      const int row = c >> 2, col = (c & 3) << 3;
      *(v8h*)(dst + row * 32 + col) = *(const v8h*)(src + (size_t)row * DD + col);
    }
#endif
  };

  stage(0);
  for (int it = 0; it < 64; ++it) {
    if (it + 1 < 64) stage(it + 1);
#if HAVE_TDM
    if (wave == 0) {
      if (it + 1 < 64) __builtin_amdgcn_s_wait_tensorcnt(1);  // cur tile done
      else             __builtin_amdgcn_s_wait_tensorcnt(0);
    }
#endif
    __syncthreads();  // publish LDS tile to all waves

    const size_t aoff = (it >> 5) ? (size_t)3 * SZA : (size_t)0;  // int select
    const _Float16* Ap = Abase + aoff + (size_t)m0 * DD + ((it & 31) << 5);
    __builtin_prefetch(Ap + 256, 0, 1);
    const v16h a0 = load_frag16(Ap, DD, lane);
    const v16h a1 = load_frag16(Ap + (size_t)16 * DD, DD, lane);
    const _Float16* Bp = &bbuf[it & 1][0];
#pragma unroll
    for (int t = 0; t < 4; ++t) {
      const v16h bf = load_frag16(Bp + t * 16 * 32, 32, lane);
      acc[0][t] = wmma16(a0, bf, acc[0][t]);
      acc[1][t] = wmma16(a1, bf, acc[1][t]);
    }
    __syncthreads();  // all waves done with this buffer before it is re-filled
  }

  // Epilogue: C layout — VGPR r, lane&15 = col; lane>=16 => rows 8..15
#pragma unroll
  for (int rf = 0; rf < 2; ++rf) {
#pragma unroll
    for (int t = 0; t < 4; ++t) {
#pragma unroll
      for (int r = 0; r < 8; ++r) {
        const int m = m0 + rf * 16 + ((lane >> 4) << 3) + r;
        const int n = bn + t * 16 + (lane & 15);
        const int b = m >> 11, l = m & (LL - 1);
        const int h = n >> 6, dk = n & 63;
        const size_t bh = (size_t)b * HH + h;
        const _Float16 v = (_Float16)acc[rf][t][r];
        if (z == 0)      Qo[(bh * LL + l) * DKV + dk] = v;
        else if (z == 1) Ko[(bh * LL + l) * DKV + dk] = v;
        else             Vt[(bh * DKV + dk) * LL + l] = v;
      }
    }
  }
}

// ---------------------------------------------------------------------------
// 4) Flash attention: grid (L/64, B*H); block 128 (4 waves)
//    Each wave: 16 queries x full head; online softmax over 64-key tiles.
// ---------------------------------------------------------------------------
__global__ __launch_bounds__(128) void attn_kernel(
    const _Float16* __restrict__ Q, const _Float16* __restrict__ K,
    const _Float16* __restrict__ Vt, _Float16* __restrict__ O) {
  constexpr int LDP = 72;  // padded LDS row stride (halves), 144B: 16B aligned
  __shared__ _Float16 ldsP[4][16][LDP];

  const int wave  = threadIdx.x >> 5;
  const int lane  = threadIdx.x & 31;
  const int bh    = blockIdx.y;
  const int b     = bh >> 4;
  const int h     = bh & 15;
  const int qbase = blockIdx.x * 64 + wave * 16;

  const _Float16* Qb = Q + ((size_t)bh * LL + qbase) * DKV;
  const _Float16* Kb = K + (size_t)bh * LL * DKV;
  const _Float16* Vb = Vt + (size_t)bh * DKV * LL;

  const v16h aq0 = load_frag16(Qb + 0, DKV, lane);
  const v16h aq1 = load_frag16(Qb + 32, DKV, lane);

  v8f o[4] = {};
  float mrow[8], lrow[8];
#pragma unroll
  for (int r = 0; r < 8; ++r) { mrow[r] = -1e30f; lrow[r] = 0.0f; }

  for (int j0 = 0; j0 < LL; j0 += 64) {
    // S = Q @ K^T  (4 col-tiles x 2 k-steps)
    v8f s[4] = {};
    const _Float16* Kp = Kb + (size_t)j0 * DKV;
#pragma unroll
    for (int t = 0; t < 4; ++t) {
      v16h b0 = load_frag16(Kp + (size_t)(t * 16) * DKV + 0, DKV, lane);
      s[t] = wmma16(aq0, b0, s[t]);
      v16h b1 = load_frag16(Kp + (size_t)(t * 16) * DKV + 32, DKV, lane);
      s[t] = wmma16(aq1, b1, s[t]);
    }

    // Online softmax per row (rows live across 16-lane halves)
#pragma unroll
    for (int r = 0; r < 8; ++r) {
      float s0 = s[0][r] * INV_SCALE, s1 = s[1][r] * INV_SCALE;
      float s2 = s[2][r] * INV_SCALE, s3 = s[3][r] * INV_SCALE;
      float mx = fmaxf(fmaxf(s0, s1), fmaxf(s2, s3));
#pragma unroll
      for (int msk = 1; msk < 16; msk <<= 1)
        mx = fmaxf(mx, __shfl_xor(mx, msk, 32));
      const float mnew = fmaxf(mrow[r], mx);
      const float corr = __expf(mrow[r] - mnew);
      const float p0 = __expf(s0 - mnew), p1 = __expf(s1 - mnew);
      const float p2 = __expf(s2 - mnew), p3 = __expf(s3 - mnew);
      float ps = p0 + p1 + p2 + p3;
#pragma unroll
      for (int msk = 1; msk < 16; msk <<= 1) ps += __shfl_xor(ps, msk, 32);
      lrow[r] = lrow[r] * corr + ps;
      mrow[r] = mnew;
#pragma unroll
      for (int t = 0; t < 4; ++t) o[t][r] *= corr;
      const int row = r + ((lane >> 4) << 3);
      _Float16* lp = &ldsP[wave][row][lane & 15];
      lp[0] = (_Float16)p0; lp[16] = (_Float16)p1;
      lp[32] = (_Float16)p2; lp[48] = (_Float16)p3;
    }

    // O += P @ V   (per-wave LDS tile -> A fragments; V^T gives contiguous B)
    const v16h ap0 = load_frag16(&ldsP[wave][0][0], LDP, lane);
    const v16h ap1 = load_frag16(&ldsP[wave][0][32], LDP, lane);
#pragma unroll
    for (int t = 0; t < 4; ++t) {
      v16h bv0 = load_frag16(Vb + (size_t)(t * 16) * LL + j0, LL, lane);
      o[t] = wmma16(ap0, bv0, o[t]);
      v16h bv1 = load_frag16(Vb + (size_t)(t * 16) * LL + j0 + 32, LL, lane);
      o[t] = wmma16(ap1, bv1, o[t]);
    }
  }

  // Normalize and write O in [B*L, H*64] layout (A-matrix for the FC GEMM)
#pragma unroll
  for (int r = 0; r < 8; ++r) {
    const float inv = 1.0f / lrow[r];
    const int lq = qbase + ((lane >> 4) << 3) + r;
    const size_t rowoff = ((size_t)b * LL + lq) * DD + h * 64;
#pragma unroll
    for (int t = 0; t < 4; ++t)
      O[rowoff + t * 16 + (lane & 15)] = (_Float16)(o[t][r] * inv);
  }
}

// ---------------------------------------------------------------------------
// 5) FC GEMM + residual: preLN = O @ Wfc + q_a + q_s   (fp32 out)
//    grid (M/128, N/64); block 128; same TDM-staged structure as QKV GEMM
// ---------------------------------------------------------------------------
__global__ __launch_bounds__(128) void fc_gemm_kernel(
    const _Float16* __restrict__ O16, const _Float16* __restrict__ WfcT,
    const float* __restrict__ qa, const float* __restrict__ qs,
    float* __restrict__ preLN) {
  __shared__ _Float16 bbuf[2][64 * 32];

  const int bm = blockIdx.x * 128, bn = blockIdx.y * 64;
  const int wave = threadIdx.x >> 5, lane = threadIdx.x & 31;
  const int m0 = bm + wave * 32;

  v8f acc[2][4] = {};

  auto stage = [&](int it) {  // it in [0,32)
    const int k0 = it << 5;
    const _Float16* src = WfcT + (size_t)bn * DD + k0;
    _Float16* dst = &bbuf[it & 1][0];
#if HAVE_TDM
    if (wave == 0)
      tdm_load_2d((unsigned long long)(size_t)src,
                  (unsigned int)(size_t)dst, 64, 32, DD);
#else
    const int c0 = threadIdx.x * 2;
#pragma unroll
    for (int c = c0; c < c0 + 2; ++c) {
      const int row = c >> 2, col = (c & 3) << 3;
      *(v8h*)(dst + row * 32 + col) = *(const v8h*)(src + (size_t)row * DD + col);
    }
#endif
  };

  stage(0);
  for (int it = 0; it < 32; ++it) {
    if (it + 1 < 32) stage(it + 1);
#if HAVE_TDM
    if (wave == 0) {
      if (it + 1 < 32) __builtin_amdgcn_s_wait_tensorcnt(1);
      else             __builtin_amdgcn_s_wait_tensorcnt(0);
    }
#endif
    __syncthreads();

    const _Float16* Ap = O16 + (size_t)m0 * DD + (it << 5);
    __builtin_prefetch(Ap + 256, 0, 1);
    const v16h a0 = load_frag16(Ap, DD, lane);
    const v16h a1 = load_frag16(Ap + (size_t)16 * DD, DD, lane);
    const _Float16* Bp = &bbuf[it & 1][0];
#pragma unroll
    for (int t = 0; t < 4; ++t) {
      const v16h bf = load_frag16(Bp + t * 16 * 32, 32, lane);
      acc[0][t] = wmma16(a0, bf, acc[0][t]);
      acc[1][t] = wmma16(a1, bf, acc[1][t]);
    }
    __syncthreads();
  }

#pragma unroll
  for (int rf = 0; rf < 2; ++rf) {
#pragma unroll
    for (int t = 0; t < 4; ++t) {
#pragma unroll
      for (int r = 0; r < 8; ++r) {
        const int m = m0 + rf * 16 + ((lane >> 4) << 3) + r;
        const int n = bn + t * 16 + (lane & 15);
        const size_t idx = (size_t)m * DD + n;
        preLN[idx] = acc[rf][t][r] + qa[idx] + qs[idx];
      }
    }
  }
}

// ---------------------------------------------------------------------------
// 6) LayerNorm: one block per row
// ---------------------------------------------------------------------------
__global__ __launch_bounds__(256) void ln_kernel(const float* __restrict__ x,
                                                 const float* __restrict__ gamma,
                                                 const float* __restrict__ beta,
                                                 float* __restrict__ out) {
  __shared__ float sh[512];
  const size_t row = blockIdx.x;
  const float* xr = x + row * DD;
  float s = 0.0f, s2 = 0.0f;
  for (int i = threadIdx.x; i < DD; i += 256) {
    const float v = xr[i];
    s += v; s2 += v * v;
  }
  sh[threadIdx.x] = s; sh[256 + threadIdx.x] = s2;
  __syncthreads();
  for (int off = 128; off > 0; off >>= 1) {
    if (threadIdx.x < off) {
      sh[threadIdx.x] += sh[threadIdx.x + off];
      sh[256 + threadIdx.x] += sh[256 + threadIdx.x + off];
    }
    __syncthreads();
  }
  const float mu = sh[0] * (1.0f / DD);
  const float var = sh[256] * (1.0f / DD) - mu * mu;
  const float rs = rsqrtf(var + 1e-6f);
  for (int i = threadIdx.x; i < DD; i += 256)
    out[row * DD + i] = (xr[i] - mu) * rs * gamma[i] + beta[i];
}

// ---------------------------------------------------------------------------
// Host launcher
// ---------------------------------------------------------------------------
extern "C" void kernel_launch(void* const* d_in, const int* in_sizes, int n_in,
                              void* d_out, int out_size, void* d_ws, size_t ws_size,
                              hipStream_t stream) {
  (void)in_sizes; (void)n_in; (void)out_size; (void)ws_size;
  const float* acts[6];
  for (int i = 0; i < 6; ++i) acts[i] = (const float*)d_in[i];   // q_a,k_a,v_a,q_s,k_s,v_s
  const float* Ws[7];
  for (int j = 0; j < 7; ++j) Ws[j] = (const float*)d_in[6 + j]; // Wqa..Wvs, Wfc
  const float* gamma = (const float*)d_in[13];
  const float* beta  = (const float*)d_in[14];
  float* out = (float*)d_out;

  // Workspace carve-up (f16 elements)
  _Float16* act16 = (_Float16*)d_ws;            // 6 * SZA
  _Float16* w16   = act16 + 6 * SZA;            // 7 * SZW (transposed)
  _Float16* Q16   = w16 + 7 * SZW;              // SZQ
  _Float16* K16   = Q16 + SZQ;                  // SZQ
  _Float16* Vt16  = K16 + SZQ;                  // SZQ (transposed per head)
  _Float16* O16   = Vt16 + SZQ;                 // SZA
  float* preLN    = (float*)d_ws;               // reuses act16 region (dead)

  // 1) activations fp32 -> f16
  for (int i = 0; i < 6; ++i)
    cvt_act_kernel<<<SZA / (256 * 4), 256, 0, stream>>>(acts[i], act16 + (size_t)i * SZA);

  // 2) weights fp32 -> f16 transposed
  for (int j = 0; j < 7; ++j)
    cvt_w_t_kernel<<<dim3(DD / 32, DD / 32), dim3(32, 8), 0, stream>>>(Ws[j], w16 + (size_t)j * SZW);

  // 3) QKV projections (q = qa@Wqa + qs@Wqs etc.)
  qkv_gemm_kernel<<<dim3(M_TOT / 128, DD / 64, 3), 128, 0, stream>>>(act16, w16, Q16, K16, Vt16);

  // 4) attention
  attn_kernel<<<dim3(LL / 64, BB * HH), 128, 0, stream>>>(Q16, K16, Vt16, O16);

  // 5) output projection + residual
  fc_gemm_kernel<<<dim3(M_TOT / 128, DD / 64), 128, 0, stream>>>(
      O16, w16 + 6 * SZW, acts[0], acts[3], preLN);

  // 6) layernorm
  ln_kernel<<<M_TOT, 256, 0, stream>>>(preLN, gamma, beta, out);
}
